// Interaction_mlp_layer_9698036155140
// MI455X (gfx1250) — matrix operand order repaired and checked by reading, not compiled, not run on hardware
//
#include <hip/hip_runtime.h>
#include <hip/hip_bf16.h>
#include <math.h>

// ---------------------------------------------------------------------------
// MI455X (gfx1250) implementation.
// Precision: f16 WMMA (v_wmma_f32_16x16x32_f16) with f32 accumulate for all
// GEMMs; f32 everywhere else. Edge MLP dominates (520K rows x 2 x 128x128).
// Edge kernel: 4 tiles per block with W1/W2/node-table resident in LDS.
// ---------------------------------------------------------------------------

typedef __attribute__((ext_vector_type(16))) _Float16 v16h;
typedef __attribute__((ext_vector_type(8)))  float    v8f;

#define Bc   32
#define Nn   128
#define INF  64
#define HIDC 128
#define OUTC 64
#define Ec   16256            // N*(N-1)
#define EBLOCKS 1024          // 32 batches * 32 blocks (4 tiles of 128 edges each)

__device__ __forceinline__ float silu_f(float x) {
    return x / (1.0f + __expf(-x));
}

// frag-layout index helpers ---------------------------------------------------
// A-frag (16x32 f16): lane m=lane&15, g=lane>>4; halfs i<8 -> k=kb+g*8+i,
//                     i>=8 -> k=kb+16+g*8+(i-8).
// B-frag (32x16 f16): lane = g*16 + n15; halfs i<8 -> k = kb + g*8 + i,
//                     i>=8 -> k = kb + 16 + g*8 + (i-8).
// W and inter-layer h live in LDS in frag order:
//   [ (kt*NT + nt)*32 + lane ] * 16 + i   -> contiguous 32B per lane.

__device__ __forceinline__ int w_frag_idx(int k, int n, int NT) {
    int kt = k >> 5, kr = k & 31;
    int gg = (kr >> 3) & 1, ii = (kr & 7) + ((kr & 16) ? 8 : 0);
    int nt = n >> 4, nl = n & 15;
    return (((kt * NT + nt) << 5) + (gg << 4) + nl) * 16 + ii;
}

__device__ __forceinline__ int h_frag_idx(int wave, int m, int k) {
    // layout matching the A-frag read pattern for the wave's own 16 rows
    int kt = k >> 5, kr = k & 31;
    int gp = (kr >> 3) & 1;
    int ip = (kr & 7) + ((kr & 16) ? 8 : 0);
    return ((((wave << 2) + kt) << 5) + (gp << 4) + m) * 16 + ip;
}

// ---------------------------------------------------------------------------
// Kernel 1: fused edge MLP (2 layers, pre-BN) with on-the-fly SEND/RECV gather.
// grid = 1024: blockIdx = b*32 + qb, block handles tiles 4*qb..4*qb+3 (<127)
// of batch b, 128 edges each. block = 256 (8 waves of 16 rows).
// Weights + node table staged ONCE per block; no per-tile barriers (sH is
// per-wave private; same-wave LDS ordering via DScnt).
// ---------------------------------------------------------------------------
__global__ __launch_bounds__(256)
void edge_mlp_kernel(const float* __restrict__ nodef,
                     const float* __restrict__ eW1, const float* __restrict__ eb1,
                     const float* __restrict__ eW2, const float* __restrict__ eb2,
                     float* __restrict__ mij, float* __restrict__ part)
{
    __shared__ _Float16 sW1f[16384];      // 32KB eW1, frag layout
    __shared__ _Float16 sW2f[16384];      // 32KB eW2, frag layout
    __shared__ _Float16 sNode[128 * 72];  // 18KB node table (pitch 72)
    __shared__ _Float16 sH[16384];        // 32KB inter-layer h, per-wave slices
    __shared__ float    sStage[4096];     // 16KB stats reduce stage

    const int tid  = threadIdx.x;
    const int lane = tid & 31;
    const int wave = tid >> 5;
    const int g    = lane >> 4;
    const int n15  = lane & 15;

    const int b  = blockIdx.x >> 5;
    const int qb = blockIdx.x & 31;

    // stage node_feature[b] (128x64 f32 -> f16)
    for (int idx = tid; idx < 128 * 64; idx += 256) {
        int r = idx >> 6, c = idx & 63;
        sNode[r * 72 + c] = (_Float16)nodef[((size_t)b * 128 + r) * 64 + c];
    }
    // stage eW1 / eW2 (128x128) into frag layout
    for (int idx = tid; idx < 128 * 128; idx += 256) {
        int k = idx >> 7, n = idx & 127;
        int fi = w_frag_idx(k, n, 8);
        sW1f[fi] = (_Float16)eW1[idx];
        sW2f[fi] = (_Float16)eW2[idx];
    }
    __syncthreads();

    // hoist biases into registers (uniform per lane across tiles)
    float bv1[8], bv2[8];
#pragma unroll
    for (int nt = 0; nt < 8; nt++) {
        bv1[nt] = eb1[(nt << 4) + n15];
        bv2[nt] = eb2[(nt << 4) + n15];
    }

    float tsum[8], tsq[8];
#pragma unroll
    for (int nt = 0; nt < 8; nt++) { tsum[nt] = 0.f; tsq[nt] = 0.f; }

    for (int tt = 0; tt < 4; tt++) {
        const int tile = (qb << 2) + tt;
        if (tile >= 127) break;
        const int e0 = tile << 7;

        // per-lane edge -> (send, recv)
        const int mrow = (wave << 4) + n15;
        const int e    = e0 + mrow;
        const int recv = e / 127;
        const int jj   = e - recv * 127;
        const int send = jj + (jj >= recv ? 1 : 0);

        v8f acc[8];
#pragma unroll
        for (int nt = 0; nt < 8; nt++) acc[nt] = (v8f){0,0,0,0,0,0,0,0};

        // layer 1: ef(128) @ eW1, ef cols [0,64)=send row, [64,128)=recv row
#pragma unroll
        for (int kt = 0; kt < 4; kt++) {
            const int srow = (kt < 2) ? send : recv;
            const _Float16* p = sNode + srow * 72 + ((kt & 1) << 5);
            v16h a;
            __builtin_memcpy(&a, p + (g << 3), 16);
            __builtin_memcpy((char*)&a + 16, p + 16 + (g << 3), 16);
#pragma unroll
            for (int nt = 0; nt < 8; nt++) {
                v16h bb;
                __builtin_memcpy(&bb, sW1f + ((((kt << 3) + nt) << 5) + lane) * 16, 32);
                acc[nt] = __builtin_amdgcn_wmma_f32_16x16x32_f16(
                    false, a, false, bb, (short)0, acc[nt], false, false);
            }
        }

        // h = silu(acc + eb1) -> own sH slice (per-wave private, no barrier)
#pragma unroll
        for (int nt = 0; nt < 8; nt++) {
            const int n = (nt << 4) + n15;
#pragma unroll
            for (int j = 0; j < 8; j++) {
                float s = silu_f(acc[nt][j] + bv1[nt]);
                int m = j + (g << 3);
                sH[h_frag_idx(wave, m, n)] = (_Float16)s;
            }
        }

        v8f acc2[8];
#pragma unroll
        for (int nt = 0; nt < 8; nt++) acc2[nt] = (v8f){0,0,0,0,0,0,0,0};

        // layer 2: h(128) @ eW2
#pragma unroll
        for (int kt = 0; kt < 4; kt++) {
            v16h a;
            __builtin_memcpy(&a, sH + ((((wave << 2) + kt) << 5) + lane) * 16, 32);
#pragma unroll
            for (int nt = 0; nt < 8; nt++) {
                v16h bb;
                __builtin_memcpy(&bb, sW2f + ((((kt << 3) + nt) << 5) + lane) * 16, 32);
                acc2[nt] = __builtin_amdgcn_wmma_f32_16x16x32_f16(
                    false, a, false, bb, (short)0, acc2[nt], false, false);
            }
        }

        // epilogue: pre-BN value -> mij region; accumulate stats in registers
#pragma unroll
        for (int nt = 0; nt < 8; nt++) {
            const int ch = (nt << 4) + n15;
            float lsum = 0.f, lsq = 0.f;
#pragma unroll
            for (int j = 0; j < 8; j++) {
                float s = silu_f(acc2[nt][j] + bv2[nt]);
                int m = j + (g << 3);
                int erow = e0 + (wave << 4) + m;
                mij[((size_t)b * Ec + erow) * 128 + ch] = s;
                lsum += s; lsq += s * s;
            }
            tsum[nt] += lsum; tsq[nt] += lsq;
        }
    }

    // deterministic per-channel stats reduce over 16 (wave,g) groups
#pragma unroll
    for (int nt = 0; nt < 8; nt++) {
        int sidx = (((wave << 1) + g) * 8 + nt) * 16 + n15;
        sStage[sidx]        = tsum[nt];
        sStage[2048 + sidx] = tsq[nt];
    }
    __syncthreads();
    {
        const int half = tid >> 7;          // 0 = sum, 1 = sumsq
        const int ch   = tid & 127;
        const float* sp = sStage + half * 2048;
        float v = 0.f;
#pragma unroll
        for (int wg = 0; wg < 16; wg++)
            v += sp[(wg * 8 + (ch >> 4)) * 16 + (ch & 15)];
        part[(size_t)blockIdx.x * 256 + tid] = v;
    }
}

// ---------------------------------------------------------------------------
// Kernel: per-channel mean/rstd from tile partials.  grid = C, block = 256.
// stat[c] = mean, stat[C+c] = rsqrt(var + 1e-5)
// ---------------------------------------------------------------------------
__global__ __launch_bounds__(256)
void stats_kernel(const float* __restrict__ part, float* __restrict__ stat,
                  int ntiles, int C, float inv)
{
    __shared__ float rs[256], rq[256];
    const int c = blockIdx.x, t = threadIdx.x;
    float s = 0.f, q = 0.f;
    for (int i = t; i < ntiles; i += 256) {
        s += part[(size_t)i * 2 * C + c];
        q += part[(size_t)i * 2 * C + C + c];
    }
    rs[t] = s; rq[t] = q; __syncthreads();
    for (int off = 128; off; off >>= 1) {
        if (t < off) { rs[t] += rs[t + off]; rq[t] += rq[t + off]; }
        __syncthreads();
    }
    if (t == 0) {
        float m = rs[0] * inv;
        float v = rq[0] * inv - m * m;
        stat[c] = m;
        stat[C + c] = rsqrtf(v + 1e-5f);
    }
}

// ---------------------------------------------------------------------------
// Kernel: normalize mij in place (BN*g+b then *em*em) and aggregate the 127
// contiguous edges of each receiver -> incoming (= sum/128). No atomics.
// grid = B*N, block = 128 (one thread per channel).
// ---------------------------------------------------------------------------
__global__ __launch_bounds__(128)
void edge_norm_agg_kernel(float* __restrict__ mij,
                          const float* __restrict__ emask,
                          const float* __restrict__ eg, const float* __restrict__ ebt,
                          const float* __restrict__ statE,
                          float* __restrict__ incoming)
{
    const int b  = blockIdx.x >> 7;
    const int r  = blockIdx.x & 127;
    const int ch = threadIdx.x;
    const float mean = statE[ch], rstd = statE[128 + ch];
    const float gv = eg[ch], bv = ebt[ch];
    float acc = 0.f;
    const int ebase = r * 127;
    for (int j = 0; j < 127; j++) {
        const int e = ebase + j;
        const float em = emask[(size_t)b * Ec + e];
        const size_t idx = ((size_t)b * Ec + e) * 128 + ch;
        float x = mij[idx];
        float y = ((x - mean) * rstd * gv + bv) * em * em;  // msg*em then *em
        mij[idx] = y;            // mij_m output
        acc += y;                // scatter-add contribution (out == y)
    }
    incoming[((size_t)b * 128 + r) * 128 + ch] = acc * (1.0f / 128.0f);
}

// ---------------------------------------------------------------------------
// Generic single-layer WMMA GEMM: out = silu(A @ W + bias), optional stats.
// A: M x K (f32), W: K x N (f32). grid = M/128, block = 256 (8 waves).
// K chunked by 64 through LDS; W staged in frag layout (conflict-free b128s).
// ---------------------------------------------------------------------------
template<int K, int N>
__global__ __launch_bounds__(256)
void gemm_silu_kernel(const float* __restrict__ A, const float* __restrict__ W,
                      const float* __restrict__ bias, float* __restrict__ out,
                      float* __restrict__ part)
{
    constexpr int NT  = N / 16;
    constexpr int KCH = K / 64;
    __shared__ _Float16 sA[128 * 72];
    __shared__ _Float16 sW[2 * NT * 32 * 16];
    __shared__ float sStage[512 * NT];

    const int tid  = threadIdx.x;
    const int lane = tid & 31;
    const int wave = tid >> 5;
    const int g    = lane >> 4;
    const int n15  = lane & 15;
    const int rows0 = blockIdx.x * 128;

    v8f acc[NT];
#pragma unroll
    for (int nt = 0; nt < NT; nt++) acc[nt] = (v8f){0,0,0,0,0,0,0,0};

    for (int kc = 0; kc < KCH; kc++) {
        if (kc) __syncthreads();
        for (int idx = tid; idx < 128 * 64; idx += 256) {
            int r = idx >> 6, c = idx & 63;
            sA[r * 72 + c] = (_Float16)A[(size_t)(rows0 + r) * K + kc * 64 + c];
        }
        for (int idx = tid; idx < 64 * N; idx += 256) {
            int k = idx / N, n = idx - k * N;
            sW[w_frag_idx(k, n, NT)] = (_Float16)W[(size_t)(kc * 64 + k) * N + n];
        }
        __syncthreads();
#pragma unroll
        for (int kt = 0; kt < 2; kt++) {
            const _Float16* p = sA + ((wave << 4) + n15) * 72 + (kt << 5);
            v16h a;
            __builtin_memcpy(&a, p + (g << 3), 16);
            __builtin_memcpy((char*)&a + 16, p + 16 + (g << 3), 16);
#pragma unroll
            for (int nt = 0; nt < NT; nt++) {
                v16h bb;
                __builtin_memcpy(&bb, sW + (((kt * NT + nt) << 5) + lane) * 16, 32);
                acc[nt] = __builtin_amdgcn_wmma_f32_16x16x32_f16(
                    false, a, false, bb, (short)0, acc[nt], false, false);
            }
        }
    }

#pragma unroll
    for (int nt = 0; nt < NT; nt++) {
        const int ch = (nt << 4) + n15;
        const float bv = bias[ch];
        float lsum = 0.f, lsq = 0.f;
#pragma unroll
        for (int j = 0; j < 8; j++) {
            float s = silu_f(acc[nt][j] + bv);
            int row = rows0 + (wave << 4) + j + (g << 3);
            out[(size_t)row * N + ch] = s;
            lsum += s; lsq += s * s;
        }
        if (part) {
            int sidx = (((wave << 1) + g) * NT + nt) * 16 + n15;
            sStage[sidx]            = lsum;
            sStage[256 * NT + sidx] = lsq;
        }
    }
    if (part) {
        __syncthreads();
        if (tid < 2 * N) {
            const int half = tid / N;
            const int ch   = tid - half * N;
            const float* sp = sStage + half * 256 * NT;
            float v = 0.f;
#pragma unroll
            for (int wg = 0; wg < 16; wg++)
                v += sp[(wg * NT + (ch >> 4)) * 16 + (ch & 15)];
            part[(size_t)blockIdx.x * (2 * N) + tid] = v;
        }
    }
}

// ---------------------------------------------------------------------------
// Build nef = [node_feature | BN(nodeH)*ng+nbt then *nodes_mask]  (4096 x 192)
// ---------------------------------------------------------------------------
__global__ __launch_bounds__(256)
void build_nef_kernel(const float* __restrict__ nodef, const float* __restrict__ nodeH,
                      const float* __restrict__ statN,
                      const float* __restrict__ ng, const float* __restrict__ nbt,
                      const float* __restrict__ nmask, float* __restrict__ nef)
{
    const int idx = blockIdx.x * 256 + threadIdx.x;
    if (idx >= 4096 * 192) return;
    const int row = idx / 192;
    const int c   = idx - row * 192;
    float v;
    if (c < 64) {
        v = nodef[(size_t)row * 64 + c];
    } else {
        const int ch = c - 64;
        float x = nodeH[(size_t)row * 128 + ch];
        v = ((x - statN[ch]) * statN[128 + ch] * ng[ch] + nbt[ch]) * nmask[row];
    }
    nef[(size_t)row * 192 + c] = v;
}

// ---------------------------------------------------------------------------
// Final BN + nodes_mask -> out_node (4096 x 64)
// ---------------------------------------------------------------------------
__global__ __launch_bounds__(256)
void final_norm_kernel(const float* __restrict__ finalH, const float* __restrict__ statM,
                       const float* __restrict__ mg, const float* __restrict__ mbt,
                       const float* __restrict__ nmask, float* __restrict__ out_node)
{
    const int idx = blockIdx.x * 256 + threadIdx.x;
    if (idx >= 4096 * 64) return;
    const int row = idx >> 6;
    const int c   = idx & 63;
    float x = finalH[idx];
    out_node[idx] = ((x - statM[c]) * statM[64 + c] * mg[c] + mbt[c]) * nmask[row];
}

// ---------------------------------------------------------------------------
extern "C" void kernel_launch(void* const* d_in, const int* in_sizes, int n_in,
                              void* d_out, int out_size, void* d_ws, size_t ws_size,
                              hipStream_t stream)
{
    const float* nodef = (const float*)d_in[0];
    const float* nmask = (const float*)d_in[1];
    const float* emask = (const float*)d_in[2];
    const float* eW1 = (const float*)d_in[3];
    const float* eb1 = (const float*)d_in[4];
    const float* eW2 = (const float*)d_in[5];
    const float* eb2 = (const float*)d_in[6];
    const float* eg  = (const float*)d_in[7];
    const float* ebt = (const float*)d_in[8];
    const float* nW1 = (const float*)d_in[9];
    const float* nb1 = (const float*)d_in[10];
    const float* nW2 = (const float*)d_in[11];
    const float* nb2 = (const float*)d_in[12];
    const float* ng  = (const float*)d_in[13];
    const float* nbt = (const float*)d_in[14];
    const float* mW1 = (const float*)d_in[15];
    const float* mb1 = (const float*)d_in[16];
    const float* mW2 = (const float*)d_in[17];
    const float* mb2 = (const float*)d_in[18];
    const float* mg  = (const float*)d_in[19];
    const float* mbt = (const float*)d_in[20];

    float* out_node = (float*)d_out;
    float* mij = out_node + (size_t)Bc * Nn * OUTC;   // second output region

    float* ws       = (float*)d_ws;
    float* partE    = ws;                          // 1024*256
    float* statE    = partE + (size_t)EBLOCKS * 256;
    float* incoming = statE + 256;                 // 4096*128
    float* hn1      = incoming + 4096 * 128;
    float* nodeH    = hn1 + 4096 * 128;
    float* partN    = nodeH + 4096 * 128;          // 32*256
    float* statN    = partN + 32 * 256;
    float* nef      = statN + 256;                 // 4096*192
    float* hm1      = nef + 4096 * 192;
    float* finalH   = hm1 + 4096 * 128;            // 4096*64
    float* partM    = finalH + 4096 * 64;          // 32*128
    float* statM    = partM + 32 * 128;

    // 1. edge MLP (pre-BN) -> mij region + stats partials
    edge_mlp_kernel<<<EBLOCKS, 256, 0, stream>>>(nodef, eW1, eb1, eW2, eb2, mij, partE);
    // 2. edge BN stats over B*E rows
    stats_kernel<<<128, 256, 0, stream>>>(partE, statE, EBLOCKS, 128, 1.0f / ((float)Bc * (float)Ec));
    // 3. normalize in place + aggregate per receiver (/N)
    edge_norm_agg_kernel<<<Bc * Nn, 128, 0, stream>>>(mij, emask, eg, ebt, statE, incoming);
    // 4-5. node MLP (two layers), layer2 emits stats partials
    gemm_silu_kernel<128, 128><<<32, 256, 0, stream>>>(incoming, nW1, nb1, hn1, nullptr);
    gemm_silu_kernel<128, 128><<<32, 256, 0, stream>>>(hn1, nW2, nb2, nodeH, partN);
    // 6. node BN stats
    stats_kernel<<<128, 256, 0, stream>>>(partN, statN, 32, 128, 1.0f / 4096.0f);
    // 7. nef = [node_feature | BN(nodeH)*mask]
    build_nef_kernel<<<3072, 256, 0, stream>>>(nodef, nodeH, statN, ng, nbt, nmask, nef);
    // 8-9. final MLP (K=192 then K=128,N=64)
    gemm_silu_kernel<192, 128><<<32, 256, 0, stream>>>(nef, mW1, mb1, hm1, nullptr);
    gemm_silu_kernel<128, 64><<<32, 256, 0, stream>>>(hm1, mW2, mb2, finalH, partM);
    // 10. final BN stats (C=64)
    stats_kernel<<<64, 256, 0, stream>>>(partM, statM, 32, 64, 1.0f / 4096.0f);
    // 11. BN + mask -> out_node
    final_norm_kernel<<<1024, 256, 0, stream>>>(finalH, statM, mg, mbt, nmask, out_node);

    (void)in_sizes; (void)n_in; (void)out_size; (void)ws_size;
    (void)nmask; (void)mb2;
}
